// RPN_79989470920812
// MI455X (gfx1250) — matrix operand order repaired and checked by reading, not compiled, not run on hardware
//
#include <hip/hip_runtime.h>
#include <hip/hip_bf16.h>
#include <math.h>

#define N_ANCH   16384
#define ROI_N    3
#define FC_HID   256
#define NMS_OUT  1024
#define NMS_THRF 0.8f
#define X_MIN_F  (-40.0f)
#define Z_MAX_F  70.0f
#define IMG_H    360
#define IMG_W    1200
#define BEV_H    700
#define BEV_W    800

typedef __attribute__((ext_vector_type(16))) _Float16 v16h;
typedef __attribute__((ext_vector_type(8)))  _Float16 v8h;
typedef __attribute__((ext_vector_type(8)))  float    v8f;

// ---------------------------------------------------------------------------
// 1x1 conv (Cin=32 -> 1) + ReLU.  Pure bandwidth: float4 loads, prefetch.
// ---------------------------------------------------------------------------
__global__ __launch_bounds__(256) void k_conv1x1_relu(
    const float* __restrict__ in, const float* __restrict__ w,
    const float* __restrict__ b, float* __restrict__ out, int npix)
{
  int p = blockIdx.x * blockDim.x + threadIdx.x;
  if (p >= npix) return;
  __builtin_prefetch((const void*)(in + (size_t)(p + 4096) * 32), 0, 1);
  const float4* v  = (const float4*)(in + (size_t)p * 32);
  const float4* wv = (const float4*)w;
  float s = 0.f;
#pragma unroll
  for (int i = 0; i < 8; ++i) {
    float4 a = v[i], ww = wv[i];
    s += a.x * ww.x + a.y * ww.y + a.z * ww.z + a.w * ww.w;
  }
  s += b[0];
  out[p] = s > 0.f ? s : 0.f;
}

// ---------------------------------------------------------------------------
// TF-style crop_and_resize (3x3 bilinear) on both maps + mask-weighted fuse.
// ---------------------------------------------------------------------------
__device__ __forceinline__ float bilin(const float* __restrict__ m, int H, int W,
                                       float y1, float x1, float y2, float x2,
                                       float fy, float fx)
{
  float ys = (y1 + fy * (y2 - y1)) * (float)(H - 1);
  float xs = (x1 + fx * (x2 - x1)) * (float)(W - 1);
  float y0 = floorf(ys), x0 = floorf(xs);
  float wy = ys - y0,    wx = xs - x0;
  int y0i = min(max((int)y0, 0), H - 1);
  int x0i = min(max((int)x0, 0), W - 1);
  int y1i = min(y0i + 1, H - 1);
  int x1i = min(x0i + 1, W - 1);
  float v00 = m[y0i * W + x0i], v01 = m[y0i * W + x1i];
  float v10 = m[y1i * W + x0i], v11 = m[y1i * W + x1i];
  return v00 * (1.f - wy) * (1.f - wx) + v01 * (1.f - wy) * wx
       + v10 * wy * (1.f - wx)         + v11 * wy * wx;
}

__global__ __launch_bounds__(256) void k_crop_fuse(
    const float* __restrict__ x1map, const float* __restrict__ x2map,
    const float* __restrict__ anchors_img, const float* __restrict__ anchors_bev,
    const float* __restrict__ img_mask, const float* __restrict__ bev_mask,
    float* __restrict__ fused)
{
  int id = blockIdx.x * blockDim.x + threadIdx.x;
  if (id >= N_ANCH * ROI_N * ROI_N) return;
  int n = id / (ROI_N * ROI_N), c = id % (ROI_N * ROI_N);
  int gy = c / ROI_N, gx = c % ROI_N;
  const float fg[3] = {0.f, 0.5f, 1.f};
  float fy = fg[gy], fx = fg[gx];
  const float* ai = anchors_img + n * 5;   // boxes = cols [2,1,4,3] = (y1,x1,y2,x2)
  const float* ab = anchors_bev + n * 5;
  float si = bilin(x1map, IMG_H, IMG_W, ai[2], ai[1], ai[4], ai[3], fy, fx);
  float sb = bilin(x2map, BEV_H, BEV_W, ab[2], ab[1], ab[4], ab[3], fy, fx);
  float mi = img_mask[0], mb = bev_mask[0];
  fused[id] = (mi * si + mb * sb) / (mi + mb);
}

// ---------------------------------------------------------------------------
// Fused FC stack with WMMA (wave32, V_WMMA_F32_16X16X32_F16).
// 4 waves/block, one 16-anchor tile per wave.  All WMMA operands staged in
// LDS (pre-swizzled into per-lane fragment layout) so the inner loops are
// pure ds_load_b128 -> v_wmma -> ds_store, with no predicated global loads.
//   GEMM1: h[16,256] = relu(A[16,32(K=9)] x fc1_w + b)   (LDS-resident h)
//   GEMM2: out[16,8] = h[16,256] x [obj_w | off_w | 0]   (8 accumulating WMMAs)
// ---------------------------------------------------------------------------
__global__ __launch_bounds__(128) void k_fc_wmma(
    const float* __restrict__ fused,
    const float* __restrict__ fc1_w, const float* __restrict__ fc1_b,
    const float* __restrict__ obj_w, const float* __restrict__ obj_b,
    const float* __restrict__ off_w, const float* __restrict__ off_b,
    float* __restrict__ objn, float* __restrict__ offs)
{
  __shared__ __align__(32) _Float16 sA[4][16][32];       //  4 KB  A tiles, K padded 9->32
  __shared__ __align__(32) _Float16 sB1[16][32][16];     // 16 KB  fc1_w per-lane B frags
  __shared__ __align__(32) _Float16 sB2[8][32][16];      //  8 KB  [obj|off|0] B frags
  __shared__ float                  sBias[FC_HID];       //  1 KB
  __shared__ __align__(32) _Float16 hbuf[4][16][FC_HID]; // 32 KB  hidden acts
  const int tid = threadIdx.x;

  // ---- zero-fill pads (vectorized v8h stores)
  {
    v8h z = {};
    v8h* pa = (v8h*)sA;
    for (int e = tid; e < 512; e += 128) pa[e] = z;      // 4096 halves
    v8h* p1 = (v8h*)sB1;
    for (int e = tid; e < 1024; e += 128) p1[e] = z;     // 8192 halves
    v8h* p2 = (v8h*)sB2;
    for (int e = tid; e < 512; e += 128) p2[e] = z;      // 4096 halves
  }
  __syncthreads();

  // ---- cooperative fills (unconditional global loads, once per block)
  for (int e = tid; e < 64 * 9; e += 128) {              // A: 64 anchors x 9 feats
    int a = e / 9, k = e % 9;
    int anchor = blockIdx.x * 64 + a;
    sA[a >> 4][a & 15][k] = (_Float16)fused[anchor * 9 + k];
  }
  for (int e = tid; e < 16 * 16 * 9; e += 128) {         // B1: K 0-8 in table order
    int j = e / 144, r = e % 144, nn = r / 9, i = r % 9;
    sB1[j][nn + (i >> 3) * 16][i & 7] = (_Float16)fc1_w[i * FC_HID + j * 16 + nn];
  }
  for (int e = tid; e < 8 * 2 * 8 * 16; e += 128) {      // B2: cols 0-7 valid
    int kk = e >> 8, r = e & 255;
    int kh = r >> 7, r2 = r & 127, nn = r2 >> 4, i = r2 & 15;
    int k = kk * 32 + (i >> 3) * 16 + kh * 8 + (i & 7);
    float w = (nn < 2) ? obj_w[k * 2 + nn] : off_w[k * 6 + (nn - 2)];
    sB2[kk][kh * 16 + nn][i] = (_Float16)w;
  }
  for (int e = tid; e < FC_HID; e += 128) sBias[e] = fc1_b[e];
  __syncthreads();

  const int lane  = tid & 31;
  const int wave  = tid >> 5;
  const int tile  = blockIdx.x * 4 + wave;
  const int m     = lane & 15;                  // A row id
  const int khalf = lane >> 4;                  // K half selected by lane group
  const int n     = lane & 15;                  // B/D column id

  // ---- A fragment: two contiguous 16B LDS reads (ds_load_b128)
  v16h a;
  {
    const v8h* pr = (const v8h*)(&sA[wave][m][0]);
    v8h lo = pr[khalf];           // K = khalf*8 .. +7
    v8h hi = pr[2 + khalf];       // K = 16 + khalf*8 .. +7
#pragma unroll
    for (int i = 0; i < 8; ++i) { a[i] = lo[i]; a[i + 8] = hi[i]; }
  }

  // ---- GEMM1 over 16 column chunks
  for (int j = 0; j < 16; ++j) {
    v16h b = ((const v16h*)sB1)[j * 32 + lane];      // contiguous 32B frag
    v8f c = {};
    c = __builtin_amdgcn_wmma_f32_16x16x32_f16(false, a, false, b,
                                               (short)0, c, false, false);
    float bias = sBias[j * 16 + n];
#pragma unroll
    for (int r = 0; r < 8; ++r) {                    // D row = khalf*8+r, col = n
      float h = c[r] + bias;
      hbuf[wave][khalf * 8 + r][j * 16 + n] = (_Float16)(h > 0.f ? h : 0.f);
    }
  }
  __syncthreads();

  // ---- GEMM2: K=256 in 8 accumulating WMMAs
  v8f c2 = {};
  for (int kk = 0; kk < 8; ++kk) {
    v16h a2;
    const v8h* hp = (const v8h*)(&hbuf[wave][m][0]);
    v8h lo = hp[kk * 4 + khalf];                     // K = kk*32 + khalf*8
    v8h hi = hp[kk * 4 + 2 + khalf];                 // K = kk*32 + 16 + khalf*8
#pragma unroll
    for (int i = 0; i < 8; ++i) { a2[i] = lo[i]; a2[i + 8] = hi[i]; }
    v16h b2 = ((const v16h*)sB2)[kk * 32 + lane];
    c2 = __builtin_amdgcn_wmma_f32_16x16x32_f16(false, a2, false, b2,
                                                (short)0, c2, false, false);
  }

  if (n < 8) {
#pragma unroll
    for (int r = 0; r < 8; ++r) {
      int anchor = tile * 16 + khalf * 8 + r;
      float v = c2[r];
      if (n < 2) objn[anchor * 2 + n]       = v + obj_b[n];
      else       offs[anchor * 6 + (n - 2)] = v + off_b[n - 2];
    }
  }
}

// ---------------------------------------------------------------------------
// Anchor regression + BEV boxes + softmax score. One thread per anchor.
// ---------------------------------------------------------------------------
__global__ __launch_bounds__(256) void k_postproc(
    const float* __restrict__ fa, const float* __restrict__ offs,
    const float* __restrict__ objn,
    float* __restrict__ reg, float* __restrict__ bev, float* __restrict__ scores)
{
  int nn = blockIdx.x * blockDim.x + threadIdx.x;
  if (nn >= N_ANCH) return;
  const float* a = fa + nn * 6;
  const float* o = offs + nn * 6;
  float p0 = o[0] * a[3] + a[0];
  float p1 = o[1] * a[4] + a[1];
  float p2 = o[2] * a[5] + a[2];
  float d0 = expf(o[3]) * a[3];
  float d1 = expf(o[4]) * a[4];
  float d2 = expf(o[5]) * a[5];
  reg[nn * 6 + 0] = p0; reg[nn * 6 + 1] = p1; reg[nn * 6 + 2] = p2;
  reg[nn * 6 + 3] = d0; reg[nn * 6 + 4] = d1; reg[nn * 6 + 5] = d2;
  float x = p0, z = p2, dx = d0, dz = d2;
  bev[nn * 4 + 0] = Z_MAX_F - (z + dz * 0.5f);
  bev[nn * 4 + 1] = x - dx * 0.5f - X_MIN_F;
  bev[nn * 4 + 2] = Z_MAX_F - (z - dz * 0.5f);
  bev[nn * 4 + 3] = x + dx * 0.5f - X_MIN_F;
  float o0 = objn[nn * 2], o1 = objn[nn * 2 + 1];
  float mx = fmaxf(o0, o1);
  float e0 = expf(o0 - mx), e1 = expf(o1 - mx);
  scores[nn] = e1 / (e0 + e1);
}

// ---------------------------------------------------------------------------
// Full 16384-element bitonic argsort (descending) entirely in LDS (128 KB).
// Also gathers sorted BEV boxes for the NMS kernel.
// ---------------------------------------------------------------------------
__global__ __launch_bounds__(1024) void k_sort(
    const float* __restrict__ scores, const float* __restrict__ bev,
    int* __restrict__ order, float* __restrict__ bsort)
{
  extern __shared__ char smem[];
  float* skey = (float*)smem;                    // 64 KB
  int*   sidx = (int*)(smem + N_ANCH * 4);       // 64 KB
  int tid = threadIdx.x;
  for (int i = tid; i < N_ANCH; i += 1024) { skey[i] = scores[i]; sidx[i] = i; }
  __syncthreads();
  for (unsigned k = 2; k <= (unsigned)N_ANCH; k <<= 1) {
    for (unsigned j = k >> 1; j > 0; j >>= 1) {
      for (unsigned i = tid; i < (unsigned)N_ANCH; i += 1024) {
        unsigned ixj = i ^ j;
        if (ixj > i) {
          bool descend = ((i & k) == 0);
          float ki = skey[i], kj = skey[ixj];
          bool sw = descend ? (ki < kj) : (ki > kj);
          if (sw) {
            skey[i] = kj; skey[ixj] = ki;
            int t = sidx[i]; sidx[i] = sidx[ixj]; sidx[ixj] = t;
          }
        }
      }
      __syncthreads();
    }
  }
  for (int i = tid; i < N_ANCH; i += 1024) {
    int oi = sidx[i];
    order[i] = oi;
    ((float4*)bsort)[i] = ((const float4*)bev)[oi];
  }
}

// ---------------------------------------------------------------------------
// Greedy NMS, 1024 sequential rounds, one workgroup; sorted box table
// (256 KB) + active bitmask live in LDS (needs CDNA5's 320 KB/WGP).
// ---------------------------------------------------------------------------
__global__ __launch_bounds__(1024) void k_nms(
    const float* __restrict__ bsort, const int* __restrict__ order,
    int* __restrict__ sel)
{
  extern __shared__ char smem[];
  float4*   sbox  = (float4*)smem;                              // 256 KB
  unsigned* amask = (unsigned*)(smem + (size_t)N_ANCH * 16);    // 2 KB
  int*      red   = (int*)(smem + (size_t)N_ANCH * 16 + 2048);
  int tid = threadIdx.x;
  for (int i = tid; i < N_ANCH; i += 1024) sbox[i] = ((const float4*)bsort)[i];
  if (tid < 512) amask[tid] = 0xFFFFFFFFu;
  __syncthreads();

  for (int it = 0; it < NMS_OUT; ++it) {
    if (tid == 0) *red = N_ANCH;
    __syncthreads();
    if (tid < 512) {
      unsigned w = amask[tid];
      if (w) atomicMin(red, tid * 32 + (__ffs((int)w) - 1));
    }
    __syncthreads();
    int j = *red;
    if (j < N_ANCH) {
      if (tid == 0) {
        sel[it] = order[j];
        atomicAnd(&amask[j >> 5], ~(1u << (j & 31)));
      }
      float4 bj = sbox[j];
      float aj = (bj.z - bj.x) * (bj.w - bj.y);
      unsigned supp = 0;
      int base = tid * 16;
#pragma unroll 4
      for (int q = 0; q < 16; ++q) {
        float4 bb = sbox[base + q];
        float yy1 = fmaxf(bj.x, bb.x), xx1 = fmaxf(bj.y, bb.y);
        float yy2 = fminf(bj.z, bb.z), xx2 = fminf(bj.w, bb.w);
        float inter = fmaxf(yy2 - yy1, 0.f) * fmaxf(xx2 - xx1, 0.f);
        float ab = (bb.z - bb.x) * (bb.w - bb.y);
        float iou = inter / (aj + ab - inter + 1e-8f);
        if (iou > NMS_THRF) supp |= (1u << q);
      }
      unsigned wm = (tid & 1) ? (supp << 16) : supp;
      if (wm) atomicAnd(&amask[tid >> 1], ~wm);
    } else {
      if (tid == 0) sel[it] = 0;
    }
    __syncthreads();
  }
}

// ---------------------------------------------------------------------------
// Output gather: d_out = [reg_anchors[top] (1024x6) | objectness[top] (1024x2)]
// ---------------------------------------------------------------------------
__global__ __launch_bounds__(256) void k_gather(
    const float* __restrict__ reg, const float* __restrict__ objn,
    const int* __restrict__ sel, float* __restrict__ out)
{
  int t = blockIdx.x * blockDim.x + threadIdx.x;
  if (t < NMS_OUT * 6) {
    int i = t / 6, c = t % 6;
    out[t] = reg[sel[i] * 6 + c];
  } else if (t < NMS_OUT * 8) {
    int tt = t - NMS_OUT * 6;
    int i = tt / 2, c = tt % 2;
    out[NMS_OUT * 6 + tt] = objn[sel[i] * 2 + c];
  }
}

// ---------------------------------------------------------------------------
extern "C" void kernel_launch(void* const* d_in, const int* in_sizes, int n_in,
                              void* d_out, int out_size, void* d_ws, size_t ws_size,
                              hipStream_t stream)
{
  const float* img_map     = (const float*)d_in[0];
  const float* bev_map     = (const float*)d_in[1];
  const float* anchors_img = (const float*)d_in[2];
  const float* anchors_bev = (const float*)d_in[3];
  const float* filtered    = (const float*)d_in[4];
  const float* img_mask    = (const float*)d_in[5];
  const float* bev_mask    = (const float*)d_in[6];
  const float* w_img       = (const float*)d_in[7];
  const float* b_img       = (const float*)d_in[8];
  const float* w_bev       = (const float*)d_in[9];
  const float* b_bev       = (const float*)d_in[10];
  const float* fc1_w       = (const float*)d_in[11];
  const float* fc1_b       = (const float*)d_in[12];
  const float* obj_w       = (const float*)d_in[13];
  const float* obj_b       = (const float*)d_in[14];
  const float* off_w       = (const float*)d_in[15];
  const float* off_b       = (const float*)d_in[16];

  const int IMG_PIX = IMG_H * IMG_W;   // 432000
  const int BEV_PIX = BEV_H * BEV_W;   // 560000

  float* ws     = (float*)d_ws;
  float* x1     = ws;                         // 432000
  float* x2     = x1 + IMG_PIX;               // 560000
  float* fused  = x2 + BEV_PIX;               // 147456
  float* objn   = fused + N_ANCH * 9;         // 32768
  float* offs   = objn + N_ANCH * 2;          // 98304
  float* reg    = offs + N_ANCH * 6;          // 98304
  float* bev    = reg + N_ANCH * 6;           // 65536
  float* scores = bev + N_ANCH * 4;           // 16384
  int*   order  = (int*)(scores + N_ANCH);    // 16384
  float* bsort  = (float*)(order + N_ANCH);   // 65536
  int*   sel    = (int*)(bsort + N_ANCH * 4); // 1024

  k_conv1x1_relu<<<(IMG_PIX + 255) / 256, 256, 0, stream>>>(img_map, w_img, b_img, x1, IMG_PIX);
  k_conv1x1_relu<<<(BEV_PIX + 255) / 256, 256, 0, stream>>>(bev_map, w_bev, b_bev, x2, BEV_PIX);
  k_crop_fuse<<<(N_ANCH * 9 + 255) / 256, 256, 0, stream>>>(
      x1, x2, anchors_img, anchors_bev, img_mask, bev_mask, fused);
  k_fc_wmma<<<N_ANCH / 64, 128, 0, stream>>>(
      fused, fc1_w, fc1_b, obj_w, obj_b, off_w, off_b, objn, offs);
  k_postproc<<<N_ANCH / 256, 256, 0, stream>>>(filtered, offs, objn, reg, bev, scores);
  k_sort<<<1, 1024, (size_t)N_ANCH * 8, stream>>>(scores, bev, order, bsort);
  k_nms<<<1, 1024, (size_t)N_ANCH * 16 + 2048 + 16, stream>>>(bsort, order, sel);
  k_gather<<<(NMS_OUT * 8 + 255) / 256, 256, 0, stream>>>(reg, objn, sel, (float*)d_out);
}
